// ExpertGroup_19224273617628
// MI455X (gfx1250) — matrix operand order repaired and checked by reading, not compiled, not run on hardware
//
#include <hip/hip_runtime.h>
#include <math.h>

// ---------------------------------------------------------------------------
// Problem constants (match reference)
// ---------------------------------------------------------------------------
#define N_EMBD   1024
#define HIDDEN   4096
#define ADAPT    256
#define NEXP     8
#define TOKENS   (4 * 2048)          // B*S = 8192
#define LN_EPS   1e-5f

typedef __bf16 bf16;
typedef __attribute__((ext_vector_type(16))) __bf16 v16bf;
typedef __attribute__((ext_vector_type(8)))  float  v8f;
typedef int v4i __attribute__((vector_size(16)));   // matches builtin param type

// ---------------------------------------------------------------------------
// CDNA5 async global->LDS copy (GLOBAL_LOAD_ASYNC_TO_LDS_B128, ASYNCcnt).
// Probe-derived signature: (v4i AS1* gsrc, v4i AS3* lds, imm offset, imm cpol).
// ---------------------------------------------------------------------------
#if defined(__has_builtin)
#  if __has_builtin(__builtin_amdgcn_global_load_async_to_lds_b128)
#    define HAVE_ASYNC_B128 1
#  endif
#  if __has_builtin(__builtin_amdgcn_s_wait_asynccnt)
#    define HAVE_WAIT_ASYNCCNT 1
#  endif
#endif
#ifndef HAVE_ASYNC_B128
#  define HAVE_ASYNC_B128 0
#endif
#ifndef HAVE_WAIT_ASYNCCNT
#  define HAVE_WAIT_ASYNCCNT 0
#endif

typedef __attribute__((address_space(1))) v4i* gv4i_ptr;
typedef __attribute__((address_space(3))) v4i* lv4i_ptr;

__device__ __forceinline__ void copy16_g2l(const bf16* __restrict__ g,
                                           bf16* __restrict__ l) {
#if HAVE_ASYNC_B128
    __builtin_amdgcn_global_load_async_to_lds_b128(
        (gv4i_ptr)(void*)const_cast<bf16*>(g),
        (lv4i_ptr)(void*)l,
        /*offset=*/0, /*cpol=*/0);
#else
    *(uint4*)l = *(const uint4*)g;   // sync fallback (ds_store path)
#endif
}

template <int N>
__device__ __forceinline__ void wait_async_le() {
#if HAVE_ASYNC_B128
#  if HAVE_WAIT_ASYNCCNT
    __builtin_amdgcn_s_wait_asynccnt(N);
#  else
    if (N == 0) asm volatile("s_wait_asynccnt 0x0" ::: "memory");
    else        asm volatile("s_wait_asynccnt 0x4" ::: "memory");
#  endif
#endif
}

// ---------------------------------------------------------------------------
// f32 -> bf16 conversion (x and weights; tiny vs GEMM cost)
// ---------------------------------------------------------------------------
__global__ void __launch_bounds__(256) cvt_f32_bf16(const float* __restrict__ in,
                                                    bf16* __restrict__ out, int n) {
    int i = blockIdx.x * 256 + threadIdx.x;
    if (i < n) out[i] = (bf16)in[i];
}

// ---------------------------------------------------------------------------
// Tiled WMMA GEMM:  C[M,N] = epilogue( A[M,K] @ W[N,K]^T ), A/W bf16 row-major.
// WG: 256 threads = 8 waves; tile 128(M) x 128(N); wave (wm,wn) -> 64x32 out
// (4x2 accumulators of V_WMMA_F32_16X16X32_BF16). K-step 32.
// Double-buffered LDS, async global->LDS copies: tile i+1 streams in under
// the WMMAs of tile i; s_wait_asynccnt 4 = "my 4 newest copies may still fly".
// MODE 0: store bf16(silu(acc))
// MODE 1: store f32(acc)
// MODE 2: store bf16(resid[idx] + 0.1f*acc)   (resid aliases Cout: in-place)
// ---------------------------------------------------------------------------
#define BM 128
#define BN 128
#define BK 32
#define LDT 40   // padded LDS row stride (bf16); 80B rows keep 16B chunk align

template <int MODE>
__global__ void __launch_bounds__(256)
gemm_wmma(const bf16* __restrict__ A, const bf16* __restrict__ W,
          void* __restrict__ Cout, const bf16* __restrict__ resid,
          int M, int N, int K)
{
    __shared__ alignas(16) bf16 As[2][BM * LDT];
    __shared__ alignas(16) bf16 Bs[2][BN * LDT];

    const int tid   = threadIdx.x;
    const int lane  = tid & 31;          // wave32
    const int wave  = tid >> 5;          // 0..7
    const int wm    = wave & 1;          // 2 waves along M (64 rows each)
    const int wn    = wave >> 1;         // 4 waves along N (32 cols each)
    const int khalf = lane >> 4;         // 0/1: K-half select (contiguous k = 16*khalf + j)
    const int l16   = lane & 15;

    const int blockM = blockIdx.y * BM;
    const int blockN = blockIdx.x * BN;

    // staging geometry: 512 16B chunks per tile; thread owns chunks tid, tid+256
    const int srow0 = tid >> 2;                  // rows 0..63
    const int scol  = (tid & 3) * 8;             // element col 0/8/16/24
    const int srow1 = srow0 + 64;                // rows 64..127

    v8f zero{};
    v8f acc[4][2];
#pragma unroll
    for (int mt = 0; mt < 4; ++mt)
#pragma unroll
        for (int nt = 0; nt < 2; ++nt) acc[mt][nt] = zero;

    auto stage_tile = [&](int buf, int k0) {
        copy16_g2l(A + (size_t)(blockM + srow0) * K + k0 + scol,
                   &As[buf][srow0 * LDT + scol]);
        copy16_g2l(A + (size_t)(blockM + srow1) * K + k0 + scol,
                   &As[buf][srow1 * LDT + scol]);
        copy16_g2l(W + (size_t)(blockN + srow0) * K + k0 + scol,
                   &Bs[buf][srow0 * LDT + scol]);
        copy16_g2l(W + (size_t)(blockN + srow1) * K + k0 + scol,
                   &Bs[buf][srow1 * LDT + scol]);
    };

    const int steps = K / BK;
    stage_tile(0, 0);

    for (int i = 0; i < steps; ++i) {
        const int cur = i & 1;
        const bool more = (i + 1) < steps;
        if (more) stage_tile(cur ^ 1, (i + 1) * BK);

        // my 4 copies for tile i are complete once <= 4 newer ones remain
        if (more) wait_async_le<4>(); else wait_async_le<0>();
        __syncthreads();                 // all waves' tile-i data visible

        // ---- B fragments (32x16): lane l16 = column n, khalf = K-half
        v16bf bfrag[2];
#pragma unroll
        for (int nt = 0; nt < 2; ++nt) {
            const bf16* bp = &Bs[cur][(wn * 32 + nt * 16 + l16) * LDT + khalf * 16];
#pragma unroll
            for (int j = 0; j < 16; ++j) bfrag[nt][j] = bp[j];
        }

        // ---- A fragments + WMMA
#pragma unroll
        for (int mt = 0; mt < 4; ++mt) {
            const bf16* ap = &As[cur][(wm * 64 + mt * 16 + l16) * LDT];
            v16bf afrag;
            // A 16x32 layout (matches B): elems 0..7  -> K = 16*khalf + j
            //                            elems 8..15 -> K = 16*khalf + 8 + j
#pragma unroll
            for (int j = 0; j < 8; ++j) afrag[j]     = ap[khalf * 16 + j];
#pragma unroll
            for (int j = 0; j < 8; ++j) afrag[8 + j] = ap[khalf * 16 + 8 + j];
#pragma unroll
            for (int nt = 0; nt < 2; ++nt) {
                acc[mt][nt] = __builtin_amdgcn_wmma_f32_16x16x32_bf16(
                    /*neg_a=*/false, afrag, /*neg_b=*/false, bfrag[nt],
                    /*c_mod=*/(short)0, acc[mt][nt],
                    /*reuse_a=*/false, /*reuse_b=*/false);
            }
        }
        __syncthreads();                 // done reading buf[cur]; next iter may
                                         // stream into the other buffer freely
    }

    // ---- epilogue: C/D layout: VGPR r -> M = r + 8*khalf, N = l16
#pragma unroll
    for (int mt = 0; mt < 4; ++mt) {
#pragma unroll
        for (int nt = 0; nt < 2; ++nt) {
            const int n = blockN + wn * 32 + nt * 16 + l16;
#pragma unroll
            for (int r = 0; r < 8; ++r) {
                const int m = blockM + wm * 64 + mt * 16 + khalf * 8 + r;
                const size_t idx = (size_t)m * N + n;
                const float v = acc[mt][nt][r];
                if (MODE == 0) {            // silu -> bf16
                    const float s = v / (1.0f + __expf(-v));
                    ((bf16*)Cout)[idx] = (bf16)s;
                } else if (MODE == 1) {     // plain f32
                    ((float*)Cout)[idx] = v;
                } else {                    // h += 0.1*acc  (bf16, in-place)
                    const float r0 = (float)resid[idx];
                    ((bf16*)Cout)[idx] = (bf16)(r0 + 0.1f * v);
                }
            }
        }
    }
}

// ---------------------------------------------------------------------------
// Per-token expert adapter + per-expert LayerNorm.
// One 256-thread block per token; W_exp_adapters (2 MB) is L2-resident.
// ---------------------------------------------------------------------------
__global__ void __launch_bounds__(256)
adapter_ln_kernel(const float* __restrict__ a,      // [T, ADAPT]
                  const float* __restrict__ ew,     // [T, NEXP]
                  const float* __restrict__ Wexp,   // [NEXP, ADAPT, ADAPT]
                  const float* __restrict__ gamma,  // [NEXP, ADAPT]
                  const float* __restrict__ beta,   // [NEXP, ADAPT]
                  bf16* __restrict__ anorm)         // [T, ADAPT]
{
    __shared__ float sa[ADAPT];
    __shared__ float red[ADAPT];
    __shared__ float sew[NEXP];

    const int t = blockIdx.x;
    const int o = threadIdx.x;

    sa[o] = a[(size_t)t * ADAPT + o];
    if (o < NEXP) sew[o] = ew[(size_t)t * NEXP + o];
    __syncthreads();

    // torch scatter-overwrite semantics: last active expert wins
    int e = -1;
#pragma unroll
    for (int i = NEXP - 1; i >= 0; --i) {
        if (e < 0 && sew[i] > 0.0f) e = i;
    }

    if (e < 0) {  // no active expert -> adapter branch contributes zero
        anorm[(size_t)t * ADAPT + o] = (bf16)0.0f;
        return;
    }

    const float* wr = Wexp + ((size_t)e * ADAPT + o) * ADAPT;
    float s = 0.0f;
#pragma unroll 8
    for (int d = 0; d < ADAPT; ++d) s = fmaf(wr[d], sa[d], s);

    // mean
    red[o] = s;
    __syncthreads();
#pragma unroll
    for (int st = 128; st > 0; st >>= 1) {
        if (o < st) red[o] += red[o + st];
        __syncthreads();
    }
    const float mu = red[0] * (1.0f / ADAPT);
    __syncthreads();

    // variance
    const float c = s - mu;
    red[o] = c * c;
    __syncthreads();
#pragma unroll
    for (int st = 128; st > 0; st >>= 1) {
        if (o < st) red[o] += red[o + st];
        __syncthreads();
    }
    const float var = red[0] * (1.0f / ADAPT);

    const float y = c * __frsqrt_rn(var + LN_EPS) * gamma[(size_t)e * ADAPT + o]
                  + beta[(size_t)e * ADAPT + o];
    anorm[(size_t)t * ADAPT + o] = (bf16)y;
}

// ---------------------------------------------------------------------------
// Host-side launch
// ---------------------------------------------------------------------------
extern "C" void kernel_launch(void* const* d_in, const int* in_sizes, int n_in,
                              void* d_out, int out_size, void* d_ws, size_t ws_size,
                              hipStream_t stream) {
    const float* x      = (const float*)d_in[0];  // [T, D]
    const float* ew     = (const float*)d_in[1];  // [T, E]
    const float* Wup    = (const float*)d_in[2];  // [H, D]
    const float* Wadapt = (const float*)d_in[3];  // [A, H]
    const float* Wexp   = (const float*)d_in[4];  // [E, A, A]
    const float* gamma  = (const float*)d_in[5];  // [E, A]
    const float* beta   = (const float*)d_in[6];  // [E, A]
    const float* Wproj  = (const float*)d_in[7];  // [H, A]
    const float* Wout   = (const float*)d_in[8];  // [D, H]
    float* out = (float*)d_out;                   // [T, D] f32

    // Workspace layout (bytes), ~112 MB total
    char* ws = (char*)d_ws;
    const size_t MB = 1024 * 1024;
    bf16*  Wup_bf    = (bf16*)(ws + 0 * MB);      // 8 MB
    bf16*  Wout_bf   = (bf16*)(ws + 8 * MB);      // 8 MB
    bf16*  Wadapt_bf = (bf16*)(ws + 16 * MB);     // 2 MB
    bf16*  Wproj_bf  = (bf16*)(ws + 18 * MB);     // 2 MB
    bf16*  h_bf      = (bf16*)(ws + 20 * MB);     // 64 MB  [T, H]
    float* a_f32     = (float*)(ws + 84 * MB);    // 8 MB   [T, A]
    bf16*  anorm_bf  = (bf16*)(ws + 92 * MB);     // 4 MB   [T, A]
    bf16*  x_bf      = (bf16*)(ws + 96 * MB);     // 16 MB  [T, D]

    // --- convert x and weights to bf16
    {
        const int n0 = TOKENS * N_EMBD;   // x
        const int n1 = HIDDEN * N_EMBD;   // W_up
        const int n2 = N_EMBD * HIDDEN;   // W_out
        const int n3 = ADAPT * HIDDEN;    // W_adapt
        const int n4 = HIDDEN * ADAPT;    // W_proj
        cvt_f32_bf16<<<(n0 + 255) / 256, 256, 0, stream>>>(x,      x_bf,      n0);
        cvt_f32_bf16<<<(n1 + 255) / 256, 256, 0, stream>>>(Wup,    Wup_bf,    n1);
        cvt_f32_bf16<<<(n2 + 255) / 256, 256, 0, stream>>>(Wout,   Wout_bf,   n2);
        cvt_f32_bf16<<<(n3 + 255) / 256, 256, 0, stream>>>(Wadapt, Wadapt_bf, n3);
        cvt_f32_bf16<<<(n4 + 255) / 256, 256, 0, stream>>>(Wproj,  Wproj_bf,  n4);
    }

    // --- 1) h = silu(x @ W_up^T)           [T, H] bf16
    gemm_wmma<0><<<dim3(HIDDEN / BN, TOKENS / BM), 256, 0, stream>>>(
        x_bf, Wup_bf, h_bf, nullptr, TOKENS, HIDDEN, N_EMBD);

    // --- 2) a = h @ W_adapt^T              [T, A] f32
    gemm_wmma<1><<<dim3(ADAPT / BN, TOKENS / BM), 256, 0, stream>>>(
        h_bf, Wadapt_bf, a_f32, nullptr, TOKENS, ADAPT, HIDDEN);

    // --- 3) per-token expert adapter + LayerNorm -> anorm  [T, A] bf16
    adapter_ln_kernel<<<TOKENS, 256, 0, stream>>>(a_f32, ew, Wexp, gamma, beta,
                                                  anorm_bf);

    // --- 4) h += 0.1 * (anorm @ W_proj^T)  (in-place; fuses the two W_out GEMMs)
    gemm_wmma<2><<<dim3(HIDDEN / BN, TOKENS / BM), 256, 0, stream>>>(
        anorm_bf, Wproj_bf, h_bf, h_bf, TOKENS, HIDDEN, ADAPT);

    // --- 5) out = (h + 0.1*h2) @ W_out^T   [T, D] f32
    gemm_wmma<1><<<dim3(N_EMBD / BN, TOKENS / BM), 256, 0, stream>>>(
        h_bf, Wout_bf, out, nullptr, TOKENS, N_EMBD, HIDDEN);
}
